// CNNS4Baseline_48223892799537
// MI455X (gfx1250) — matrix operand order repaired and checked by reading, not compiled, not run on hardware
//
#include <hip/hip_runtime.h>
#include <hip/hip_bf16.h>
#include <math.h>

// MI455X / gfx1250 implementation.
// Heavy contractions (conv2 implicit-GEMM, S4D causal Toeplitz conv, GLU
// projection) use v_wmma_f32_16x16x32_bf16 (wave32 WMMA, fp32 accumulate).
// Fragment layouts arranged for b128 loads: transposed LDS tiles, reversed+
// zero-padded Toeplitz kernel row (causal mask for free), transposed
// projection/head weights, transposed conv1 input. S4 waves compute both
// b-tiles per l-tile so the scalar Toeplitz gather is shared by 2 WMMAs.

typedef __attribute__((ext_vector_type(16))) __bf16 v16bf;
typedef __attribute__((ext_vector_type(8)))  float  v8f;

#define Bq 32
#define Lq 2048
#define Hq 256
#define DINq 51
#define KREVN 2176   // 2048 reversed + 128 zero pad

// ---------------------------------------------------------------- prep: weight reshapes
__global__ __launch_bounds__(256) void prep_weights(const float* conv2_w, const float* glu_w,
                                                    const float* head_w, __bf16* w2bf,
                                                    __bf16* gwT, float* hwT) {
  int idx = blockIdx.x * 256 + threadIdx.x;
  const int NW2 = Hq * Hq * 3;          // 196608
  const int NGW = Hq * 2 * Hq;          // 131072
  const int NHW = DINq * Hq;            // 13056
  if (idx < NW2) {
    int ho = idx / (Hq * 3);
    int k  = idx % (Hq * 3);
    int dk = k / Hq, hi = k % Hq;
    w2bf[idx] = (__bf16)conv2_w[(ho * Hq + hi) * 3 + dk];
  } else if (idx < NW2 + NGW) {
    int e = idx - NW2;
    int j = e >> 8, k = e & 255;                      // gwT[j][k] = glu_w[k][j]
    gwT[e] = (__bf16)glu_w[k * 512 + j];
  } else if (idx < NW2 + NGW + NHW) {
    int e = idx - NW2 - NGW;
    int o = e >> 8, hh = e & 255;                     // hwT[o][hh] = head_w[hh][o]
    hwT[e] = head_w[hh * DINq + o];
  }
}

// ---------------------------------------------------------------- x transpose (B,L,51)->(B,51,L)
__global__ __launch_bounds__(256) void transpose_x(const float* x, float* xT) {
  int idx = blockIdx.x * 256 + threadIdx.x;           // (b*51+d)*2048 + l
  int l  = idx & 2047;
  int bd = idx >> 11;
  int d = bd % DINq, b = bd / DINq;
  xT[idx] = x[((size_t)b * Lq + l) * DINq + d];       // coalesced writes
}

// ---------------------------------------------------------------- S4D kernel generation
__global__ __launch_bounds__(256) void s4d_gen(const float* log_dt, const float* Alr,
                                               const float* Aim, const float* Cre,
                                               const float* Cim, __bf16* krev) {
  __shared__ float ctr[64], cti[64], dre[64], dimv[64];
  int h = blockIdx.x >> 3;              // 8 l-blocks of 256 per h
  int lblk = blockIdx.x & 7;
  int tid = threadIdx.x;
  if (tid < 64) {
    float dt  = __expf(log_dt[h]);
    float are = -__expf(Alr[h * 64 + tid]);
    float aim = Aim[h * 64 + tid];
    float dr = are * dt, di = aim * dt;
    float er = __expf(dr), s, c;
    __sincosf(di, &s, &c);
    float nre = er * c - 1.0f, nim = er * s;          // exp(dtA) - 1
    float den = are * are + aim * aim;
    float qre = (nre * are + nim * aim) / den;        // (exp(dtA)-1)/A
    float qim = (nim * are - nre * aim) / den;
    float cr = Cre[h * 64 + tid], ci = Cim[h * 64 + tid];
    ctr[tid] = cr * qre - ci * qim;
    cti[tid] = cr * qim + ci * qre;
    dre[tid] = dr; dimv[tid] = di;
  }
  __syncthreads();
  int l = lblk * 256 + tid;
  float fl = (float)l;
  float acc = 0.0f;
  #pragma unroll 4
  for (int n = 0; n < 64; ++n) {
    float er = __expf(dre[n] * fl);
    float s, c;
    __sincosf(dimv[n] * fl, &s, &c);
    acc += ctr[n] * er * c - cti[n] * er * s;         // Re(Ct * exp(dtA*l))
  }
  krev[(size_t)h * KREVN + (2047 - l)] = (__bf16)(2.0f * acc);
  if (lblk == 0 && tid < 128) krev[(size_t)h * KREVN + 2048 + tid] = (__bf16)0.0f;
}

// ---------------------------------------------------------------- conv1 + BN + ReLU + 1x1 residual
__global__ __launch_bounds__(256) void conv1_k(const float* xT, const float* w1, const float* b1,
                                               const float* g1, const float* be1,
                                               const float* mu1, const float* va1,
                                               const float* rw, const float* rb, float* y1f) {
  int idx = blockIdx.x * 256 + threadIdx.x;           // b*H*L + ho*L + l
  int b  = idx / (Hq * Lq);
  int r0 = idx % (Hq * Lq);
  int ho = r0 / Lq;
  int l  = r0 % Lq;
  float acc = b1[ho];
  float res = rb[ho];
  const float* xb = xT + (size_t)b * DINq * Lq + l;   // element d at xb[d*Lq]
  const float* wb = w1 + (size_t)ho * DINq * 3;
  const float* rwb = rw + (size_t)ho * DINq;
  bool hasm = (l > 0), hasp = (l < Lq - 1);
  #pragma unroll 3
  for (int d = 0; d < DINq; ++d) {
    float xc = xb[d * Lq];
    float xm = hasm ? xb[d * Lq - 1] : 0.0f;
    float xp = hasp ? xb[d * Lq + 1] : 0.0f;
    acc += wb[d * 3 + 0] * xm + wb[d * 3 + 1] * xc + wb[d * 3 + 2] * xp;
    res += rwb[d] * xc;
  }
  float inv = g1[ho] * rsqrtf(va1[ho] + 1e-5f);
  float v = acc * inv + (be1[ho] - mu1[ho] * inv);
  v = fmaxf(v, 0.0f) + res;
  y1f[idx] = v;
}

// ---------------------------------------------------------------- conv2 (implicit GEMM, WMMA bf16)
__global__ __launch_bounds__(256) void conv2_wmma(const float* y1, const __bf16* w2bf,
                                                  const float* b2, const float* g2,
                                                  const float* be2, const float* mu2,
                                                  const float* va2, float* yfhbl, __bf16* ybf) {
  __shared__ __bf16 ytT[2][16][40];                   // [buf][n][k], row stride 80B
  int bid = blockIdx.x;
  int b  = bid >> 7;                                  // L/16 = 128 tiles
  int l0 = (bid & 127) << 4;
  int tid = threadIdx.x, wv = tid >> 5, lane = tid & 31;
  int m0 = wv * 32;                                   // this wave: ho tiles m0, m0+16
  int mrow = lane & 15;
  int kbA = (lane < 16) ? 0 : 16;
  int nB  = lane & 15;
  int kbB = (lane < 16) ? 0 : 16;
  int sn = tid & 15, sk = tid >> 4;                   // staging decode (512 = 16n x 32k)

  auto stage = [&](int buf, int step) {
    int dk  = step >> 3;
    int hi0 = (step & 7) << 5;
    int ls  = l0 + sn + dk - 1;
    bool ok = (ls >= 0 && ls < Lq);
    #pragma unroll
    for (int half = 0; half < 2; ++half) {
      int k = sk + half * 16;
      float v = ok ? y1[((size_t)b * Hq + hi0 + k) * Lq + ls] : 0.0f;
      ytT[buf][sn][k] = (__bf16)v;
    }
  };

  v8f acc[2] = {v8f{}, v8f{}};
  stage(0, 0);
  __syncthreads();
  for (int step = 0; step < 24; ++step) {             // K = 768 = 3 dk * 256 hi
    int cur = step & 1;
    if (step + 1 < 24) stage(cur ^ 1, step + 1);
    v16bf a0, a1, bb;
    const __bf16* pw0 = w2bf + (size_t)(m0 + mrow) * 768 + step * 32;
    const __bf16* pw1 = w2bf + (size_t)(m0 + 16 + mrow) * 768 + step * 32;
    const __bf16* pb  = &ytT[cur][nB][kbB];
    #pragma unroll
    for (int j = 0; j < 16; ++j) {
      a0[j] = pw0[kbA + j];
      a1[j] = pw1[kbA + j];
    }
    #pragma unroll
    for (int j = 0; j < 16; ++j) bb[j] = pb[j];       // contiguous, 16B-aligned
    acc[0] = __builtin_amdgcn_wmma_f32_16x16x32_bf16(false, a0, false, bb, (short)0, acc[0], false, false);
    acc[1] = __builtin_amdgcn_wmma_f32_16x16x32_bf16(false, a1, false, bb, (short)0, acc[1], false, false);
    __syncthreads();
  }
  int l  = l0 + (lane & 15);
  int rb = (lane < 16) ? 0 : 8;
  #pragma unroll
  for (int t = 0; t < 2; ++t) {
    #pragma unroll
    for (int r = 0; r < 8; ++r) {
      int ho = m0 + t * 16 + rb + r;
      float inv = g2[ho] * rsqrtf(va2[ho] + 1e-5f);
      float v = (acc[t][r] + b2[ho]) * inv + (be2[ho] - mu2[ho] * inv);
      v = fmaxf(v, 0.0f) + y1[((size_t)b * Hq + ho) * Lq + l];   // identity residual
      yfhbl[((size_t)ho * Bq + b) * Lq + l] = v;                 // (H,B,L) fp32
      ybf[((size_t)ho * Bq + b) * Lq + l] = (__bf16)v;           // (H,B,L) bf16
    }
  }
}

// ---------------------------------------------------------------- S4D causal Toeplitz conv (WMMA bf16)
__global__ __launch_bounds__(256) void s4_wmma(const __bf16* ybf, const __bf16* krevg,
                                               const float* yfhbl, const float* Ds4, __bf16* sbf) {
  __shared__ __bf16 kk[KREVN];                         // reversed kernel + 128 zero pad
  __shared__ __bf16 yt[2][32][72];                     // [buf][b][t], row stride 144B
  int h  = blockIdx.x >> 4;                            // 16 l-blocks of 128 per h
  int l0 = (blockIdx.x & 15) << 7;
  int tid = threadIdx.x, wv = tid >> 5, lane = tid & 31;
  int ln0 = l0 + wv * 16;                              // this wave's l-tile
  for (int e = tid; e < KREVN; e += 256) kk[e] = krevg[(size_t)h * KREVN + e];
  int mrow = lane & 15;
  int kbA = (lane < 16) ? 0 : 16;
  int nB  = lane & 15;
  int kbB = (lane < 16) ? 0 : 16;
  int si = tid >> 3, sc = (tid & 7) << 3;              // staging decode (2048 = 32b x 64t)

  auto stageY = [&](int buf, int t0) {
    const __bf16* src = ybf + ((size_t)h * Bq + si) * Lq + t0 + sc;
    #pragma unroll
    for (int q = 0; q < 8; ++q) yt[buf][si][sc + q] = src[q];
  };

  v8f c0 = {}, c1 = {};
  int nIt = (l0 >> 6) + 2;                             // causal t-blocks: t <= l0+127
  stageY(0, 0);
  __syncthreads();
  for (int it = 0; it < nIt; ++it) {
    int cur = it & 1;
    int t0 = it << 6;
    if (it + 1 < nIt) {
      stageY(cur ^ 1, t0 + 64);
      if (it + 2 < nIt)
        __builtin_prefetch(ybf + ((size_t)h * Bq + si) * Lq + t0 + 128 + sc, 0, 3);
    }
    #pragma unroll
    for (int s = 0; s < 2; ++s) {                      // two 32-wide k-steps per tile
      int ts = t0 + s * 32;
      v16bf a0, a1, bb;
      const __bf16* pa0 = &yt[cur][mrow][s * 32 + kbA];
      const __bf16* pa1 = &yt[cur][16 + mrow][s * 32 + kbA];
      #pragma unroll
      for (int j = 0; j < 16; ++j) {
        a0[j] = pa0[j];
        a1[j] = pa1[j];
      }
      int base = 2047 - (ln0 + nB) + ts + kbB;         // krev index: ascending in j
      const __bf16* pk = &kk[base];
      #pragma unroll
      for (int j = 0; j < 16; ++j) bb[j] = pk[j];
      c0 = __builtin_amdgcn_wmma_f32_16x16x32_bf16(false, a0, false, bb, (short)0, c0, false, false);
      c1 = __builtin_amdgcn_wmma_f32_16x16x32_bf16(false, a1, false, bb, (short)0, c1, false, false);
    }
    __syncthreads();
  }
  float dsk = Ds4[h];
  int l  = ln0 + (lane & 15);
  int rb = (lane < 16) ? 0 : 8;
  #pragma unroll
  for (int r = 0; r < 8; ++r) {
    int b0 = rb + r;
    int b1 = 16 + rb + r;
    float v0 = c0[r] + dsk * yfhbl[((size_t)h * Bq + b0) * Lq + l];
    float v1 = c1[r] + dsk * yfhbl[((size_t)h * Bq + b1) * Lq + l];
    float g0 = 0.5f * v0 * (1.0f + tanhf(0.7978845608f * (v0 + 0.044715f * v0 * v0 * v0)));
    float g1 = 0.5f * v1 * (1.0f + tanhf(0.7978845608f * (v1 + 0.044715f * v1 * v1 * v1)));
    sbf[((size_t)b0 * Lq + l) * Hq + h] = (__bf16)g0;
    sbf[((size_t)b1 * Lq + l) * Hq + h] = (__bf16)g1;
  }
}

// ---------------------------------------------------------------- GLU projection GEMM (WMMA bf16)
__global__ __launch_bounds__(256) void glu_wmma(const __bf16* sbf, const __bf16* gwT,
                                                const float* glub, float* g) {
  __shared__ __bf16 st[16][264];
  int row0 = blockIdx.x << 4;
  int tid = threadIdx.x, wv = tid >> 5, lane = tid & 31;
  for (int e = tid; e < 16 * 256; e += 256) {
    int m = e >> 8, k = e & 255;
    st[m][k] = sbf[(((size_t)(row0 + m)) << 8) + k];
  }
  __syncthreads();
  int mrow = lane & 15;
  int kbA = (lane < 16) ? 0 : 16;
  int nB  = lane & 15;
  int kbB = (lane < 16) ? 0 : 16;
  #pragma unroll
  for (int pp = 0; pp < 2; ++pp) {
    int j0 = (wv + pp * 8) << 4;                       // gate column pair (j0, j0+256)
    v8f c1 = {}, c2 = {};
    const __bf16* w1 = gwT + ((size_t)(j0 + nB)) * 256 + kbB;
    const __bf16* w2 = gwT + ((size_t)(j0 + 256 + nB)) * 256 + kbB;
    for (int k0 = 0; k0 < 256; k0 += 32) {
      v16bf a, b1, b2;
      const __bf16* pa = &st[mrow][k0 + kbA];
      #pragma unroll
      for (int j = 0; j < 16; ++j) { a[j] = pa[j]; }
      #pragma unroll
      for (int j = 0; j < 16; ++j) { b1[j] = w1[k0 + j]; b2[j] = w2[k0 + j]; }
      c1 = __builtin_amdgcn_wmma_f32_16x16x32_bf16(false, a, false, b1, (short)0, c1, false, false);
      c2 = __builtin_amdgcn_wmma_f32_16x16x32_bf16(false, a, false, b2, (short)0, c2, false, false);
    }
    int col = j0 + (lane & 15);
    int rb  = (lane < 16) ? 0 : 8;
    float bb1 = glub[col], bb2 = glub[col + 256];
    #pragma unroll
    for (int r = 0; r < 8; ++r) {
      int row = row0 + rb + r;
      float z1 = c1[r] + bb1;
      float z2 = c2[r] + bb2;
      g[(size_t)row * Hq + col] = z1 / (1.0f + __expf(-z2));     // GLU
    }
  }
}

// ---------------------------------------------------------------- LayerNorm + head
__global__ __launch_bounds__(256) void ln_head(const float* g, const float* lng, const float* lnb,
                                               const float* hwT, const float* hb, float* out) {
  __shared__ float sh[8][257];
  int tid = threadIdx.x, wv = tid >> 5, lane = tid & 31;
  size_t row = (size_t)blockIdx.x * 8 + wv;
  float vals[8];
  float sum = 0.0f, sq = 0.0f;
  #pragma unroll
  for (int cidx = 0; cidx < 8; ++cidx) {
    float v = g[row * Hq + cidx * 32 + lane];
    vals[cidx] = v; sum += v; sq += v * v;
  }
  #pragma unroll
  for (int off = 16; off >= 1; off >>= 1) {
    sum += __shfl_xor(sum, off, 32);
    sq  += __shfl_xor(sq,  off, 32);
  }
  float mu  = sum * (1.0f / 256.0f);
  float var = sq * (1.0f / 256.0f) - mu * mu;
  float inv = rsqrtf(var + 1e-5f);
  #pragma unroll
  for (int cidx = 0; cidx < 8; ++cidx) {
    int hc = cidx * 32 + lane;
    sh[wv][hc] = (vals[cidx] - mu) * inv * lng[hc] + lnb[hc];
  }
  __syncthreads();
  for (int o = lane; o < DINq; o += 32) {
    float acc = hb[o];
    const float* hp = hwT + (size_t)o * 256;           // contiguous per lane
    #pragma unroll 8
    for (int hh = 0; hh < Hq; ++hh) acc += sh[wv][hh] * hp[hh];
    out[row * DINq + o] = acc;
  }
}

// ---------------------------------------------------------------- launch
extern "C" void kernel_launch(void* const* d_in, const int* in_sizes, int n_in,
                              void* d_out, int out_size, void* d_ws, size_t ws_size,
                              hipStream_t stream) {
  const float* x       = (const float*)d_in[0];
  const float* conv1_w = (const float*)d_in[1];
  const float* conv1_b = (const float*)d_in[2];
  const float* bn1_g   = (const float*)d_in[3];
  const float* bn1_b   = (const float*)d_in[4];
  const float* bn1_m   = (const float*)d_in[5];
  const float* bn1_v   = (const float*)d_in[6];
  const float* res1_w  = (const float*)d_in[7];
  const float* res1_b  = (const float*)d_in[8];
  const float* conv2_w = (const float*)d_in[9];
  const float* conv2_b = (const float*)d_in[10];
  const float* bn2_g   = (const float*)d_in[11];
  const float* bn2_b   = (const float*)d_in[12];
  const float* bn2_m   = (const float*)d_in[13];
  const float* bn2_v   = (const float*)d_in[14];
  const float* log_dt  = (const float*)d_in[15];
  const float* A_lr    = (const float*)d_in[16];
  const float* A_im    = (const float*)d_in[17];
  const float* C_re    = (const float*)d_in[18];
  const float* C_im    = (const float*)d_in[19];
  const float* D_s4    = (const float*)d_in[20];
  const float* glu_w   = (const float*)d_in[21];
  const float* glu_b   = (const float*)d_in[22];
  const float* ln_g    = (const float*)d_in[23];
  const float* ln_b    = (const float*)d_in[24];
  const float* head_w  = (const float*)d_in[25];
  const float* head_b  = (const float*)d_in[26];
  float* out = (float*)d_out;

  char* ws = (char*)d_ws;
  float*  y1f  = (float*)(ws);                         // 64 MB  (B,H,L); reused as GLU out
  float*  yf   = (float*)(ws + (64ull << 20));         // 64 MB  (H,B,L) fp32
  __bf16* ybf  = (__bf16*)(ws + (128ull << 20));       // 32 MB  (H,B,L) bf16
  __bf16* krev = (__bf16*)(ws + (160ull << 20));       // ~1.1MB (H, 2176)
  __bf16* w2bf = (__bf16*)(ws + (162ull << 20));       // 384 KB
  __bf16* gwT  = (__bf16*)(ws + (162ull << 20) + (1ull << 19)); // 256 KB
  __bf16* sbf  = (__bf16*)(ws + (163ull << 20));       // 32 MB  (B,L,H)
  float*  xT   = (float*)(ws + (195ull << 20));        // 12.8MB (B,51,L)
  float*  hwT  = (float*)(ws + (208ull << 20));        // 52 KB  (51,256)
  float*  gbuf = y1f;                                  // GLU output (B,L,H) fp32

  prep_weights<<<1332, 256, 0, stream>>>(conv2_w, glu_w, head_w, w2bf, gwT, hwT);
  transpose_x<<<(Bq * DINq * Lq) / 256, 256, 0, stream>>>(x, xT);
  s4d_gen<<<Hq * 8, 256, 0, stream>>>(log_dt, A_lr, A_im, C_re, C_im, krev);
  conv1_k<<<(Bq * Hq * Lq) / 256, 256, 0, stream>>>(xT, conv1_w, conv1_b, bn1_g, bn1_b,
                                                    bn1_m, bn1_v, res1_w, res1_b, y1f);
  conv2_wmma<<<Bq * (Lq / 16), 256, 0, stream>>>(y1f, w2bf, conv2_b, bn2_g, bn2_b,
                                                 bn2_m, bn2_v, yf, ybf);
  s4_wmma<<<Hq * (Lq / 128), 256, 0, stream>>>(ybf, krev, yf, D_s4, sbf);
  glu_wmma<<<(Bq * Lq) / 16, 256, 0, stream>>>(sbf, gwT, glu_b, gbuf);
  ln_head<<<(Bq * Lq) / 8, 256, 0, stream>>>(gbuf, ln_g, ln_b, hwT, head_b, out);
}
